// RWKV_11347303596234
// MI455X (gfx1250) — matrix operand order repaired and checked by reading, not compiled, not run on hardware
//
#include <hip/hip_runtime.h>
#include <hip/hip_bf16.h>
#include <math.h>

// ---------------- model dims (fixed by reference) ----------------
#define T_   1024
#define C_   1024
#define L_   4
#define H_   16
#define N_   64
#define V_   32000
#define DFF_ 4096

typedef __bf16 bf16_t;
typedef __attribute__((ext_vector_type(4)))  bf16_t v4bf;
typedef __attribute__((ext_vector_type(8)))  bf16_t v8bf;
typedef __attribute__((ext_vector_type(16))) bf16_t v16bf;
typedef __attribute__((ext_vector_type(8)))  float  v8f;

// ---------------- epilogues ----------------
#define EPI_STORE   0
#define EPI_ADD     1
#define EPI_RELU2   2
#define EPI_TANH    3
#define EPI_SIGMOID 4

__device__ __forceinline__ float sigmoidf_(float x) { return 1.0f / (1.0f + __expf(-x)); }

template <int EPI>
__device__ __forceinline__ void epilogue(float* p, float v) {
    if      (EPI == EPI_STORE)   { *p = v; }
    else if (EPI == EPI_ADD)     { *p = *p + v; }
    else if (EPI == EPI_RELU2)   { float r = fmaxf(v, 0.0f); *p = r * r; }
    else if (EPI == EPI_TANH)    { *p = tanhf(v); }
    else if (EPI == EPI_SIGMOID) { *p = sigmoidf_(v); }
}

// ---------------- WMMA bf16 GEMM: C[M,N] = A[M,K] @ B[K,N] ----------------
// REQUIREMENTS (met by every call below): M % 128 == 0, K % 32 == 0, N >= 4.
// 128(M) x 128(N) tile per 256-thread WG (8 waves, 4x2 wave grid).
// Each wave: 32x64 strip -> 2 A-frags x 4 B-frags = 8 WMMAs per K-step.
// LDS double-buffered: As[2][128][32] bf16, Bs[2][128][32] bf16 (B transposed
// so fragments are contiguous-k -> ds_load_b128 pairs). 32 KB total.
//
// bf16 A 16x32 frag (ISA 7.12.2): lane l: m=l&15, hi=l>>4; elem e -> k=(e>>3)*16+hi*8+(e&7)
// bf16 B 32x16 frag:              lane l: n=l&15, hi=l>>4; elem e -> k=(e>>3)*16+hi*8+(e&7)
// f32  C/D 16x16:                 vgpr r, lane l: n=l&15, m=r+8*(l>>4)
#define BM_ 128
#define BN_ 128
#define BK_ 32

template <int EPI>
__global__ __launch_bounds__(256)
void gemm_bf16_wmma(const float* __restrict__ A, const float* __restrict__ Bm,
                    float* __restrict__ Cm, int M, int N, int K) {
    __shared__ __align__(16) bf16_t As[2][BM_][BK_];
    __shared__ __align__(16) bf16_t Bs[2][BN_][BK_];

    const int m0 = blockIdx.y * BM_;
    const int n0 = blockIdx.x * BN_;
    const int tid  = threadIdx.x;
    const int lane = tid & 31;
    const int wid  = tid >> 5;
    const int wm = wid >> 1;        // 0..3 -> 32-row M strip
    const int wn = wid & 1;         // 0..1 -> 64-col N strip
    const int hi = lane >> 4;
    const int l15 = lane & 15;

    // staging: A tile 128x32 floats = 1024 float4 slots; thread does 4.
    // slot s: m = s>>3 (row has 8 float4), kq = (s&7)*4
    // staging: B tile 32x128 floats = 1024 float4 slots; thread does 4.
    // slot s: k = s>>5 (row has 32 float4), nq = (s&31)*4
    float4 ra[4];
    float4 rb[4];

    auto gload = [&](int k0) {
        #pragma unroll
        for (int i = 0; i < 4; ++i) {
            int s = tid + i * 256;
            int m = s >> 3, kq = (s & 7) * 4;
            ra[i] = *(const float4*)(A + (size_t)(m0 + m) * K + k0 + kq);
        }
        #pragma unroll
        for (int i = 0; i < 4; ++i) {
            int s = tid + i * 256;
            int k = s >> 5;
            int nq = (s & 31) * 4;
            int gn = n0 + nq;
            gn = (gn <= N - 4) ? gn : (N - 4);   // clamp: dup columns, never stored
            rb[i] = *(const float4*)(Bm + (size_t)(k0 + k) * N + gn);
        }
    };
    auto stage = [&](int buf) {
        #pragma unroll
        for (int i = 0; i < 4; ++i) {
            int s = tid + i * 256;
            int m = s >> 3, kq = (s & 7) * 4;
            v4bf c;
            c[0] = (bf16_t)ra[i].x; c[1] = (bf16_t)ra[i].y;
            c[2] = (bf16_t)ra[i].z; c[3] = (bf16_t)ra[i].w;
            *(v4bf*)&As[buf][m][kq] = c;
        }
        #pragma unroll
        for (int i = 0; i < 4; ++i) {
            int s = tid + i * 256;
            int k = s >> 5;
            int nq = (s & 31) * 4;
            Bs[buf][nq + 0][k] = (bf16_t)rb[i].x;
            Bs[buf][nq + 1][k] = (bf16_t)rb[i].y;
            Bs[buf][nq + 2][k] = (bf16_t)rb[i].z;
            Bs[buf][nq + 3][k] = (bf16_t)rb[i].w;
        }
    };

    v8f acc[2][4] = {};

    gload(0);
    stage(0);
    int cur = 0;

    for (int k0 = 0; k0 < K; k0 += BK_) {
        __syncthreads();                       // buf[cur] visible to all waves
        const bool more = (k0 + BK_) < K;
        if (more) gload(k0 + BK_);             // hide global latency behind WMMA
        if (k0 + 2 * BK_ < K) {                // pull tile-after-next toward L2
            __builtin_prefetch(A + (size_t)(m0 + (tid >> 1)) * K + k0 + 2 * BK_, 0, 3);
            __builtin_prefetch(Bm + (size_t)(k0 + 2 * BK_ + (tid >> 3)) * N + n0, 0, 3);
        }

        v16bf af[2];
        #pragma unroll
        for (int t = 0; t < 2; ++t) {
            const int mr = wm * 32 + t * 16 + l15;
            v8bf alo = *(const v8bf*)&As[cur][mr][hi * 8];
            v8bf ahi = *(const v8bf*)&As[cur][mr][16 + hi * 8];
            af[t] = __builtin_shufflevector(alo, ahi,
                      0,1,2,3,4,5,6,7,8,9,10,11,12,13,14,15);
        }

        #pragma unroll
        for (int s = 0; s < 4; ++s) {
            const int nr = wn * 64 + s * 16 + l15;
            v8bf bl = *(const v8bf*)&Bs[cur][nr][hi * 8];
            v8bf bh = *(const v8bf*)&Bs[cur][nr][16 + hi * 8];
            v16bf bf = __builtin_shufflevector(bl, bh,
                         0,1,2,3,4,5,6,7,8,9,10,11,12,13,14,15);
            acc[0][s] = __builtin_amdgcn_wmma_f32_16x16x32_bf16(false, af[0], false, bf,
                                                                (short)0, acc[0][s], false, false);
            acc[1][s] = __builtin_amdgcn_wmma_f32_16x16x32_bf16(false, af[1], false, bf,
                                                                (short)0, acc[1][s], false, false);
        }

        if (more) {                            // write-only to the other buffer
            stage(cur ^ 1);
            cur ^= 1;
        }
    }

    #pragma unroll
    for (int t = 0; t < 2; ++t) {
        #pragma unroll
        for (int rr = 0; rr < 8; ++rr) {
            const int row = m0 + wm * 32 + t * 16 + rr + 8 * hi;
            if (row >= M) continue;
            #pragma unroll
            for (int s = 0; s < 4; ++s) {
                const int col = n0 + wn * 64 + s * 16 + l15;
                if (col < N) epilogue<EPI>(&Cm[(size_t)row * N + col], acc[t][s][rr]);
            }
        }
    }
}

// ---------------- LayerNorm over C (one block per t, 256 threads) ----------------
__global__ __launch_bounds__(256)
void ln_rows(const float* __restrict__ in, float* __restrict__ out,
             const float* __restrict__ w, const float* __restrict__ b) {
    const int t = blockIdx.x, tid = threadIdx.x;
    const float* row = in + (size_t)t * C_;
    __shared__ float red[256];
    float s = 0.0f;
    for (int c = tid; c < C_; c += 256) s += row[c];
    red[tid] = s; __syncthreads();
    for (int st = 128; st > 0; st >>= 1) { if (tid < st) red[tid] += red[tid + st]; __syncthreads(); }
    float mean = red[0] * (1.0f / C_); __syncthreads();
    float s2 = 0.0f;
    for (int c = tid; c < C_; c += 256) { float d = row[c] - mean; s2 += d * d; }
    red[tid] = s2; __syncthreads();
    for (int st = 128; st > 0; st >>= 1) { if (tid < st) red[tid] += red[tid + st]; __syncthreads(); }
    float rstd = rsqrtf(red[0] * (1.0f / C_) + 1e-5f);
    for (int c = tid; c < C_; c += 256)
        out[(size_t)t * C_ + c] = (row[c] - mean) * rstd * w[c] + b[c];
}

// ---------------- embedding gather + LN0 ----------------
__global__ __launch_bounds__(256)
void embed_ln0(const float* __restrict__ emb, const int* __restrict__ idx,
               const float* __restrict__ w, const float* __restrict__ b,
               float* __restrict__ out) {
    const int t = blockIdx.x, tid = threadIdx.x;
    const float* row = emb + (size_t)idx[t] * C_;
    __shared__ float red[256];
    float s = 0.0f;
    for (int c = tid; c < C_; c += 256) s += row[c];
    red[tid] = s; __syncthreads();
    for (int st = 128; st > 0; st >>= 1) { if (tid < st) red[tid] += red[tid + st]; __syncthreads(); }
    float mean = red[0] * (1.0f / C_); __syncthreads();
    float s2 = 0.0f;
    for (int c = tid; c < C_; c += 256) { float d = row[c] - mean; s2 += d * d; }
    red[tid] = s2; __syncthreads();
    for (int st = 128; st > 0; st >>= 1) { if (tid < st) red[tid] += red[tid + st]; __syncthreads(); }
    float rstd = rsqrtf(red[0] * (1.0f / C_) + 1e-5f);
    for (int c = tid; c < C_; c += 256)
        out[(size_t)t * C_ + c] = (row[c] - mean) * rstd * w[c] + b[c];
}

// ---------------- token-shift mixes: 6 outputs (r,w,k,v,a,g) ----------------
__global__ __launch_bounds__(256)
void mix6(const float* __restrict__ Y,
          const float* __restrict__ xr, const float* __restrict__ xw,
          const float* __restrict__ xk, const float* __restrict__ xv,
          const float* __restrict__ xa, const float* __restrict__ xg,
          float* __restrict__ M) {
    int i = blockIdx.x * 256 + threadIdx.x;
    if (i >= T_ * C_) return;
    int t = i / C_, c = i - t * C_;
    float y = Y[i];
    float d = ((t > 0) ? Y[i - C_] : 0.0f) - y;
    const size_t TC = (size_t)T_ * C_;
    M[0 * TC + i] = y + d * xr[c];
    M[1 * TC + i] = y + d * xw[c];
    M[2 * TC + i] = y + d * xk[c];
    M[3 * TC + i] = y + d * xv[c];
    M[4 * TC + i] = y + d * xa[c];
    M[5 * TC + i] = y + d * xg[c];
}

__global__ __launch_bounds__(256)
void mix1(const float* __restrict__ Y, const float* __restrict__ xm,
          float* __restrict__ M) {
    int i = blockIdx.x * 256 + threadIdx.x;
    if (i >= T_ * C_) return;
    int t = i / C_, c = i - t * C_;
    float y = Y[i];
    float d = ((t > 0) ? Y[i - C_] : 0.0f) - y;
    M[i] = y + d * xm[c];
}

// ---------------- per-head block reduce (64 threads) ----------------
__device__ __forceinline__ float reduce64(float v, float* red, int i) {
    red[i] = v; __syncthreads();
    #pragma unroll
    for (int s = 32; s > 0; s >>= 1) { if (i < s) red[i] += red[i + s]; __syncthreads(); }
    float r = red[0]; __syncthreads();
    return r;
}

// ---------------- prepare recurrence inputs ----------------
__global__ __launch_bounds__(64)
void prep_tmix(int first,
               const float* __restrict__ w0, const float* __restrict__ a0,
               const float* __restrict__ v0, const float* __restrict__ kkp,
               const float* __restrict__ kap,
               float* __restrict__ Kb, float* __restrict__ WW, float* __restrict__ AA,
               const float* __restrict__ VG, float* __restrict__ Vb,
               float* __restrict__ VF, float* __restrict__ AN, float* __restrict__ BN) {
    const int h = blockIdx.x, t = blockIdx.y, i = threadIdx.x;
    const int c = h * N_ + i;
    const size_t idx = (size_t)t * C_ + c;
    __shared__ float red[64];

    float kv = Kb[idx];

    float z = w0[c] + WW[idx];
    float wlog = -log1pf(__expf(-z)) - 0.5f;      // -softplus(-z) - 0.5
    WW[idx] = __expf(-__expf(wlog));              // decay in (0,1)

    float a = sigmoidf_(a0[c] + AA[idx]);
    AA[idx] = a;

    float vv = Vb[idx];
    if (first) {
        VF[idx] = vv;
    } else {
        float gate = sigmoidf_(v0[c] + VG[idx]);
        vv = vv + (VF[idx] - vv) * gate;
        Vb[idx] = vv;
    }

    float kk = kv * kkp[c];
    float ss = reduce64(kk * kk, red, i);
    float nrm = fmaxf(sqrtf(ss), 1e-12f);
    float kkn = kk / nrm;
    AN[idx] = -kkn;
    BN[idx] = kkn * a;

    Kb[idx] = kv * (1.0f + (a - 1.0f) * kap[c]);
}

// ---------------- RWKV7 recurrence: S = S*diag(w) + (S a) b^T + v k^T ----------------
__global__ __launch_bounds__(64)
void rwkv7_scan(const float* __restrict__ R, const float* __restrict__ W,
                const float* __restrict__ Kb, const float* __restrict__ Vb,
                const float* __restrict__ A, const float* __restrict__ Bm,
                float* __restrict__ O) {
    const int h = blockIdx.x, i = threadIdx.x;
    __shared__ float sr[64], sw[64], sk[64], sv[64], sa[64], sb[64];
    float S[64];
    #pragma unroll
    for (int j = 0; j < 64; ++j) S[j] = 0.0f;

    for (int t = 0; t < T_; ++t) {
        const size_t base = (size_t)t * C_ + h * N_;
        sr[i] = R[base + i];  sw[i] = W[base + i];  sk[i] = Kb[base + i];
        sv[i] = Vb[base + i]; sa[i] = A[base + i];  sb[i] = Bm[base + i];
        __syncthreads();

        float sav = 0.0f;
        #pragma unroll
        for (int j = 0; j < 64; ++j) sav += S[j] * sa[j];

        const float vi = sv[i];
        float out = 0.0f;
        #pragma unroll
        for (int j = 0; j < 64; ++j) {
            S[j] = S[j] * sw[j] + sav * sb[j] + vi * sk[j];
            out += S[j] * sr[j];
        }
        O[base + i] = out;
        __syncthreads();
    }
}

// ---------------- groupnorm + (r*k*r_k)-bonus + gate ----------------
__global__ __launch_bounds__(64)
void post_tmix(const float* __restrict__ RO, const float* __restrict__ R,
               const float* __restrict__ Kb, const float* __restrict__ Vb,
               const float* __restrict__ G,
               const float* __restrict__ lnxw, const float* __restrict__ lnxb,
               const float* __restrict__ rk, float* __restrict__ GT) {
    const int h = blockIdx.x, t = blockIdx.y, i = threadIdx.x;
    const int c = h * N_ + i;
    const size_t idx = (size_t)t * C_ + c;
    __shared__ float red[64];

    float o = RO[idx];
    float mean = reduce64(o, red, i) * (1.0f / N_);
    float d = o - mean;
    float var = reduce64(d * d, red, i) * (1.0f / N_);
    float gn = d * rsqrtf(var + 64e-5f) * lnxw[c] + lnxb[c];
    float dot = reduce64(R[idx] * Kb[idx] * rk[h * N_ + i], red, i);
    GT[idx] = (gn + dot * Vb[idx]) * G[idx];
}

// ---------------- host orchestration ----------------
template <int EPI>
static inline void gemm(const float* A, const float* Bm, float* Cm,
                        int M, int N, int K, hipStream_t s) {
    dim3 g((N + BN_ - 1) / BN_, (M + BM_ - 1) / BM_), b(256);
    gemm_bf16_wmma<EPI><<<g, b, 0, s>>>(A, Bm, Cm, M, N, K);
}

extern "C" void kernel_launch(void* const* d_in, const int* in_sizes, int n_in,
                              void* d_out, int out_size, void* d_ws, size_t ws_size,
                              hipStream_t stream) {
    (void)in_sizes; (void)n_in; (void)out_size; (void)ws_size;

    const float* emb     = (const float*)d_in[0];
    const float* headW   = (const float*)d_in[1];
    const float* ln0_w   = (const float*)d_in[2];
    const float* ln0_b   = (const float*)d_in[3];
    const float* lnout_w = (const float*)d_in[4];
    const float* lnout_b = (const float*)d_in[5];
    const float* ln1_w   = (const float*)d_in[6];
    const float* ln1_b   = (const float*)d_in[7];
    const float* ln2_w   = (const float*)d_in[8];
    const float* ln2_b   = (const float*)d_in[9];
    const float* lnx_w   = (const float*)d_in[10];
    const float* lnx_b   = (const float*)d_in[11];
    const float* x_r     = (const float*)d_in[12];
    const float* x_w     = (const float*)d_in[13];
    const float* x_k     = (const float*)d_in[14];
    const float* x_v     = (const float*)d_in[15];
    const float* x_a     = (const float*)d_in[16];
    const float* x_g     = (const float*)d_in[17];
    const float* w0      = (const float*)d_in[18];
    const float* a0      = (const float*)d_in[19];
    const float* v0      = (const float*)d_in[20];
    const float* k_k     = (const float*)d_in[21];
    const float* k_a     = (const float*)d_in[22];
    const float* ffn_xk  = (const float*)d_in[23];
    const float* w1      = (const float*)d_in[24];
    const float* a1      = (const float*)d_in[25];
    const float* v1      = (const float*)d_in[26];
    const float* g1      = (const float*)d_in[27];
    const float* w2      = (const float*)d_in[28];
    const float* a2      = (const float*)d_in[29];
    const float* v2      = (const float*)d_in[30];
    const float* g2      = (const float*)d_in[31];
    const float* r_k     = (const float*)d_in[32];
    const float* Wr      = (const float*)d_in[33];
    const float* Wk      = (const float*)d_in[34];
    const float* Wv      = (const float*)d_in[35];
    const float* Wo      = (const float*)d_in[36];
    const float* Wk_ffn  = (const float*)d_in[37];
    const float* Wv_ffn  = (const float*)d_in[38];
    const int*   idx     = (const int*)d_in[39];

    float* out = (float*)d_out;
    float* ws  = (float*)d_ws;

    const size_t TC = (size_t)T_ * C_;
    float* X   = ws + 0 * TC;
    float* Y   = ws + 1 * TC;
    float* MIX = ws + 2 * TC;      // 6 buffers: [2..7]
    float* M0 = MIX + 0 * TC, *M1 = MIX + 1 * TC, *M2 = MIX + 2 * TC;
    float* M3 = MIX + 3 * TC, *M4 = MIX + 4 * TC, *M5 = MIX + 5 * TC;
    float* R   = ws + 8 * TC;
    float* Kb  = ws + 9 * TC;
    float* Vb  = ws + 10 * TC;
    float* WW  = ws + 11 * TC;
    float* AA  = ws + 12 * TC;
    float* VG  = ws + 13 * TC;
    float* Gb  = ws + 14 * TC;
    float* VF  = ws + 15 * TC;
    float* AN  = ws + 16 * TC;
    float* BN  = ws + 17 * TC;
    float* RO  = ws + 18 * TC;
    float* GT  = ws + 19 * TC;
    float* LT  = ws + 20 * TC;     // LoRA stage-1 tmp (<= T*128)
    float* FB  = ws + 21 * TC;     // FFN hidden [T, DFF] = 4*TC -> [21..25)

    const int nEW = (T_ * C_ + 255) / 256;

    embed_ln0<<<T_, 256, 0, stream>>>(emb, idx, ln0_w, ln0_b, X);

    for (int i = 0; i < L_; ++i) {
        const size_t iC = (size_t)i * C_;
        const size_t iCC = (size_t)i * C_ * C_;

        ln_rows<<<T_, 256, 0, stream>>>(X, Y, ln1_w + iC, ln1_b + iC);
        mix6<<<nEW, 256, 0, stream>>>(Y, x_r + iC, x_w + iC, x_k + iC,
                                      x_v + iC, x_a + iC, x_g + iC, MIX);

        // main projections (WMMA)
        gemm<EPI_STORE>(M0, Wr + iCC, R,  T_, C_, C_, stream);
        gemm<EPI_STORE>(M2, Wk + iCC, Kb, T_, C_, C_, stream);
        gemm<EPI_STORE>(M3, Wv + iCC, Vb, T_, C_, C_, stream);

        // LoRA chains
        gemm<EPI_TANH >(M1, w1 + (size_t)i * C_ * 64,  LT, T_, 64,  C_, stream);
        gemm<EPI_STORE>(LT, w2 + (size_t)i * 64 * C_,  WW, T_, C_,  64, stream);
        gemm<EPI_STORE>(M4, a1 + (size_t)i * C_ * 64,  LT, T_, 64,  C_, stream);
        gemm<EPI_STORE>(LT, a2 + (size_t)i * 64 * C_,  AA, T_, C_,  64, stream);
        if (i > 0) {
            gemm<EPI_STORE>(M3, v1 + (size_t)i * C_ * 32, LT, T_, 32, C_, stream);
            gemm<EPI_STORE>(LT, v2 + (size_t)i * 32 * C_, VG, T_, C_, 32, stream);
        }
        gemm<EPI_SIGMOID>(M5, g1 + (size_t)i * C_ * 128, LT, T_, 128, C_, stream);
        gemm<EPI_STORE  >(LT, g2 + (size_t)i * 128 * C_, Gb, T_, C_, 128, stream);

        prep_tmix<<<dim3(H_, T_), 64, 0, stream>>>(
            (i == 0) ? 1 : 0, w0 + iC, a0 + iC, v0 + iC, k_k + iC, k_a + iC,
            Kb, WW, AA, VG, Vb, VF, AN, BN);

        rwkv7_scan<<<H_, 64, 0, stream>>>(R, WW, Kb, Vb, AN, BN, RO);

        post_tmix<<<dim3(H_, T_), 64, 0, stream>>>(
            RO, R, Kb, Vb, Gb, lnx_w + iC, lnx_b + iC, r_k + (size_t)i * C_, GT);

        gemm<EPI_ADD>(GT, Wo + iCC, X, T_, C_, C_, stream);

        // FFN
        ln_rows<<<T_, 256, 0, stream>>>(X, Y, ln2_w + iC, ln2_b + iC);
        mix1<<<nEW, 256, 0, stream>>>(Y, ffn_xk + iC, M0);
        gemm<EPI_RELU2>(M0, Wk_ffn + (size_t)i * C_ * DFF_, FB, T_, DFF_, C_, stream);
        gemm<EPI_ADD  >(FB, Wv_ffn + (size_t)i * DFF_ * C_, X,  T_, C_, DFF_, stream);
    }

    ln_rows<<<T_, 256, 0, stream>>>(X, Y, lnout_w, lnout_b);
    gemm<EPI_STORE>(Y, headW, out, T_, V_, C_, stream);
}
